// BaseEncoder_68702296867023
// MI455X (gfx1250) — matrix operand order repaired and checked by reading, not compiled, not run on hardware
//
#include <hip/hip_runtime.h>
#include <math.h>

#define BB 16
#define SS 512
#define EE 512
#define HH 8
#define DH 64
#define FF 2048
#define LL 6
#define DD 200

typedef __bf16 bf16;
typedef __attribute__((ext_vector_type(16))) __bf16 v16bf;
typedef __attribute__((ext_vector_type(8)))  __bf16 v8bf;
typedef __attribute__((ext_vector_type(8)))  float  v8f;

union Frag16 { v16bf v; v8bf h[2]; };

__device__ inline v8f zero8() {
    v8f z;
#pragma unroll
    for (int i = 0; i < 8; ++i) z[i] = 0.0f;
    return z;
}

__device__ inline v8f wmma_bf16(v16bf a, v16bf b, v8f c) {
    // 8 args: (neg_a, A, neg_b, B, c_mod, C, reuse_a, reuse_b)
    return __builtin_amdgcn_wmma_f32_16x16x32_bf16(false, a, false, b, (short)0, c, false, false);
}

// ---------------------------------------------------------------------------
// Generic GEMM: Y[M,N] = X[M,K] @ W[N,K]^T + bias, optional exact GELU.
// Block = 128 threads = 4 waves (2x2); each wave computes a 32x64 tile
// (2 M-frags x 4 N-frags -> 8 WMMAs per 12 b128 fragment loads).
// Block tile = 64 x 128.
// ---------------------------------------------------------------------------
__global__ __launch_bounds__(128) void gemm_bf16_k(
    const bf16* __restrict__ X, const bf16* __restrict__ W,
    const float* __restrict__ bias,
    float* __restrict__ outf, bf16* __restrict__ outb,
    int M, int N, int K, int act)
{
    const int lane = threadIdx.x & 31;
    const int wave = threadIdx.x >> 5;
    const int wm0  = blockIdx.y * 64  + (wave >> 1) * 32;
    const int wn0  = blockIdx.x * 128 + (wave & 1) * 64;
    const int l15  = lane & 15;
    const int hi   = (lane & 16) ? 1 : 0;

    v8f acc[2][4];
#pragma unroll
    for (int i = 0; i < 2; ++i)
#pragma unroll
        for (int j = 0; j < 4; ++j) acc[i][j] = zero8();

    const int aklo = hi * 8;    // A-frag: K chunks at +0 and +16 from here
    const int bkb  = hi * 8;    // B-frag: K chunks at +0 and +16 from here

    const bf16* arow0 = X + (size_t)(wm0 + l15) * K;
    const bf16* arow1 = X + (size_t)(wm0 + 16 + l15) * K;
    const bf16* brow[4];
#pragma unroll
    for (int j = 0; j < 4; ++j)
        brow[j] = W + (size_t)(wn0 + j * 16 + l15) * K;

    for (int k0 = 0; k0 < K; k0 += 32) {
        Frag16 a[2], bf[4];
        {
            const bf16* p0 = arow0 + k0 + aklo;
            const bf16* p1 = arow1 + k0 + aklo;
            a[0].h[0] = *(const v8bf*)p0; a[0].h[1] = *(const v8bf*)(p0 + 16);
            a[1].h[0] = *(const v8bf*)p1; a[1].h[1] = *(const v8bf*)(p1 + 16);
        }
#pragma unroll
        for (int j = 0; j < 4; ++j) {
            const bf16* p = brow[j] + k0 + bkb;
            bf[j].h[0] = *(const v8bf*)p; bf[j].h[1] = *(const v8bf*)(p + 16);
        }
        __builtin_prefetch(brow[0] + k0 + bkb + 32, 0, 3);   // -> global_prefetch_b8
        __builtin_prefetch(brow[2] + k0 + bkb + 32, 0, 3);
#pragma unroll
        for (int i = 0; i < 2; ++i)
#pragma unroll
            for (int j = 0; j < 4; ++j)
                acc[i][j] = wmma_bf16(a[i].v, bf[j].v, acc[i][j]);
    }

    const int mb = hi * 8;
#pragma unroll
    for (int i = 0; i < 2; ++i) {
#pragma unroll
        for (int j = 0; j < 4; ++j) {
            int mt = wm0 + i * 16;
            int n  = wn0 + j * 16 + l15;
            float bv = bias ? bias[n] : 0.0f;
#pragma unroll
            for (int r = 0; r < 8; ++r) {
                int m = mt + mb + r;
                float v = acc[i][j][r] + bv;
                if (act) v = 0.5f * v * (1.0f + erff(v * 0.70710678118654752f));
                if (outf) outf[(size_t)m * N + n] = v;
                if (outb) outb[(size_t)m * N + n] = (bf16)v;
            }
        }
    }
}

// ---------------------------------------------------------------------------
// Flash attention with relative-distance bias + key padding mask.
// Block = 128 threads = 4 waves; each wave owns 16 query rows of one (b,h).
// S^T = K@Q^T via WMMA (per-lane softmax stats), O^T = V^T@P^T via WMMA.
// dist_emb staged in LDS; squares/mask loads vectorized (int4/float4).
// ---------------------------------------------------------------------------
__global__ __launch_bounds__(128) void attn_flash_k(
    const bf16* __restrict__ qkv, const int* __restrict__ squares,
    const float* __restrict__ dist_emb, const float* __restrict__ maskf,
    bf16* __restrict__ ctxb)
{
    __shared__ __align__(16) bf16 vT[DH][32];       // V^T tile [d][key], shared by block
    __shared__ __align__(16) bf16 Pl[4][16][32];    // P tile [q][key], per wave
    __shared__ float distS[DD * HH];                // 6.4 KB bias table

    const int lane = threadIdx.x & 31;
    const int wave = threadIdx.x >> 5;
    const int b    = blockIdx.y / HH;
    const int h    = blockIdx.y % HH;
    const int q0   = blockIdx.x * 64 + wave * 16;
    const int qc   = lane & 15;
    const int hi   = (lane & 16) ? 1 : 0;

    for (int i = threadIdx.x; i < DD * HH; i += 128) distS[i] = dist_emb[i];

    const size_t rs = 3 * EE;  // qkv row stride (elements)
    const bf16* qbase = qkv + ((size_t)b * SS) * rs + (size_t)h * DH;
    const bf16* kbase = qbase + EE;
    const bf16* vbase = qbase + 2 * EE;

    // Q loaded once as B-fragments (n = query col, K = head dim)
    Frag16 qf0, qf1;
    {
        const bf16* p0 = qbase + (size_t)(q0 + qc) * rs + hi * 8;        // kdim 0..31
        const bf16* p1 = qbase + (size_t)(q0 + qc) * rs + 32 + hi * 8;   // kdim 32..63
        qf0.h[0] = *(const v8bf*)p0; qf0.h[1] = *(const v8bf*)(p0 + 16);
        qf1.h[0] = *(const v8bf*)p1; qf1.h[1] = *(const v8bf*)(p1 + 16);
    }

    v8f o0 = zero8(), o1 = zero8(), o2 = zero8(), o3 = zero8();
    float mrun = -3.0e38f, lrun = 0.0f;
    const float scale = 0.125f;  // 1/sqrt(64)
    const int* sqrow = squares + (size_t)b * SS * SS + (size_t)(q0 + qc) * SS;
    const float* padrow = maskf + (size_t)b * SS;

    for (int k0 = 0; k0 < SS; k0 += 32) {
        __syncthreads();  // previous iteration's LDS reads complete (also covers distS)

        // Cooperative stage of V^T into LDS: thread -> (key, 16-d chunk)
        {
            int keyl = threadIdx.x >> 2;
            int d0   = (threadIdx.x & 3) * 16;
            const bf16* src = vbase + (size_t)(k0 + keyl) * rs + d0;
            v8bf t0 = *(const v8bf*)src;
            v8bf t1 = *(const v8bf*)(src + 8);
#pragma unroll
            for (int j = 0; j < 8; ++j) {
                vT[d0 + j][keyl]     = t0[j];
                vT[d0 + 8 + j][keyl] = t1[j];
            }
        }

        // S^T = K @ Q^T for two 16-key tiles; bias + mask on the fly
        float vals[2][8];
        float mloc = -3.0e38f;
#pragma unroll
        for (int t = 0; t < 2; ++t) {
            Frag16 a0, a1;
            const bf16* ap0 = kbase + (size_t)(k0 + t * 16 + qc) * rs + hi * 8;
            const bf16* ap1 = kbase + (size_t)(k0 + t * 16 + qc) * rs + 32 + hi * 8;
            a0.h[0] = *(const v8bf*)ap0; a0.h[1] = *(const v8bf*)(ap0 + 16);
            a1.h[0] = *(const v8bf*)ap1; a1.h[1] = *(const v8bf*)(ap1 + 16);
            v8f st = zero8();
            st = wmma_bf16(a0.v, qf0.v, st);
            st = wmma_bf16(a1.v, qf1.v, st);

            // vectorized bias index / pad-mask loads (8 consecutive keys/lane)
            const int keyb = k0 + t * 16 + hi * 8;
            const int4*   sq4 = (const int4*)(sqrow + keyb);
            const float4* pm4 = (const float4*)(padrow + keyb);
            int4   s0 = sq4[0], s1 = sq4[1];
            float4 m0 = pm4[0], m1 = pm4[1];
            int   di[8]  = { s0.x, s0.y, s0.z, s0.w, s1.x, s1.y, s1.z, s1.w };
            float pmv[8] = { m0.x, m0.y, m0.z, m0.w, m1.x, m1.y, m1.z, m1.w };
#pragma unroll
            for (int r = 0; r < 8; ++r) {
                float v = st[r] * scale + distS[di[r] * HH + h] + pmv[r];
                vals[t][r] = v;
                mloc = fmaxf(mloc, v);
            }
        }

        // Online softmax (stats per query column; lane pair reduction)
        float mstrip = fmaxf(mloc, __shfl_xor(mloc, 16, 32));
        float mnew   = fmaxf(mrun, mstrip);
        float f      = __expf(mrun - mnew);
        float sloc = 0.0f;
#pragma unroll
        for (int t = 0; t < 2; ++t) {
#pragma unroll
            for (int r = 0; r < 8; ++r) {
                float p = __expf(vals[t][r] - mnew);
                sloc += p;
                Pl[wave][qc][t * 16 + hi * 8 + r] = (bf16)p;
            }
        }
        float ssum = sloc + __shfl_xor(sloc, 16, 32);
        lrun = lrun * f + ssum;
        mrun = mnew;
#pragma unroll
        for (int r = 0; r < 8; ++r) { o0[r] *= f; o1[r] *= f; o2[r] *= f; o3[r] *= f; }

        __syncthreads();  // vT and Pl visible

        // O^T += V^T @ P^T  (A from vT, B from Pl)
        Frag16 pb;
        {
            const bf16* p = &Pl[wave][qc][hi * 8];
            pb.h[0] = *(const v8bf*)p; pb.h[1] = *(const v8bf*)(p + 16);
        }
#pragma unroll
        for (int t = 0; t < 4; ++t) {
            Frag16 va;
            const bf16* p = &vT[t * 16 + qc][hi * 8];
            va.h[0] = *(const v8bf*)p; va.h[1] = *(const v8bf*)(p + 16);
            v8f& accum = (t == 0) ? o0 : (t == 1) ? o1 : (t == 2) ? o2 : o3;
            accum = wmma_bf16(va.v, pb.v, accum);
        }
    }

    // Write ctx (bf16) from O^T layout: lane = q col, regs = d rows
    float linv = 1.0f / lrun;
    bf16* dst = ctxb + ((size_t)b * SS + q0 + qc) * EE + (size_t)h * DH;
#pragma unroll
    for (int t = 0; t < 4; ++t) {
        const v8f& accum = (t == 0) ? o0 : (t == 1) ? o1 : (t == 2) ? o2 : o3;
#pragma unroll
        for (int r = 0; r < 8; ++r) {
            int d = t * 16 + hi * 8 + r;
            dst[d] = (bf16)(accum[r] * linv);
        }
    }
}

// ---------------------------------------------------------------------------
// Fused residual + LayerNorm (one block per row of E=512)
// ---------------------------------------------------------------------------
__global__ __launch_bounds__(128) void add_ln_k(
    const float* __restrict__ x, const float* __restrict__ y,
    const float* __restrict__ w, const float* __restrict__ b,
    float* __restrict__ xo, bf16* __restrict__ xob)
{
    __shared__ float rsum[128], rsq[128];
    const int row = blockIdx.x;
    const int tid = threadIdx.x;
    const float* xr = x + (size_t)row * EE;
    const float* yr = y ? (y + (size_t)row * EE) : nullptr;
    float v[4]; float s = 0.0f, q = 0.0f;
#pragma unroll
    for (int i = 0; i < 4; ++i) {
        int e = tid + i * 128;
        float t = xr[e] + (yr ? yr[e] : 0.0f);
        v[i] = t; s += t; q += t * t;
    }
    rsum[tid] = s; rsq[tid] = q;
    __syncthreads();
    for (int off = 64; off > 0; off >>= 1) {
        if (tid < off) { rsum[tid] += rsum[tid + off]; rsq[tid] += rsq[tid + off]; }
        __syncthreads();
    }
    float mean = rsum[0] * (1.0f / EE);
    float var  = rsq[0] * (1.0f / EE) - mean * mean;
    float rstd = rsqrtf(var + 1e-5f);
    float* xor_ = xo + (size_t)row * EE;
    bf16* xbr   = xob + (size_t)row * EE;
#pragma unroll
    for (int i = 0; i < 4; ++i) {
        int e = tid + i * 128;
        float t = (v[i] - mean) * rstd * w[e] + b[e];
        xor_[e] = t; xbr[e] = (bf16)t;
    }
}

// Final LayerNorm with (B,S,E) -> (S,B,E) transposed store into d_out
__global__ __launch_bounds__(128) void final_ln_k(
    const float* __restrict__ x, const float* __restrict__ w,
    const float* __restrict__ b, float* __restrict__ out)
{
    __shared__ float rsum[128], rsq[128];
    const int row = blockIdx.x;          // row = bIdx*S + s
    const int tid = threadIdx.x;
    const float* xr = x + (size_t)row * EE;
    float v[4]; float s = 0.0f, q = 0.0f;
#pragma unroll
    for (int i = 0; i < 4; ++i) {
        int e = tid + i * 128;
        float t = xr[e];
        v[i] = t; s += t; q += t * t;
    }
    rsum[tid] = s; rsq[tid] = q;
    __syncthreads();
    for (int off = 64; off > 0; off >>= 1) {
        if (tid < off) { rsum[tid] += rsum[tid + off]; rsq[tid] += rsq[tid + off]; }
        __syncthreads();
    }
    float mean = rsum[0] * (1.0f / EE);
    float var  = rsq[0] * (1.0f / EE) - mean * mean;
    float rstd = rsqrtf(var + 1e-5f);
    int bi = row / SS, si = row % SS;
    float* o = out + ((size_t)si * BB + bi) * EE;
#pragma unroll
    for (int i = 0; i < 4; ++i) {
        int e = tid + i * 128;
        o[e] = (v[i] - mean) * rstd * w[e] + b[e];
    }
}

// ---------------------------------------------------------------------------
// Embedding, pad mask, weight f32->bf16 conversion
// ---------------------------------------------------------------------------
__global__ void embed_k(const int* __restrict__ seq, const float* __restrict__ tok,
                        float* __restrict__ xf, bf16* __restrict__ xb)
{
    size_t i = (size_t)blockIdx.x * blockDim.x + threadIdx.x;
    if (i >= (size_t)BB * SS * EE) return;
    size_t bs = i / EE; int e = (int)(i % EE);
    float v = tok[(size_t)seq[bs] * EE + e] * 22.62741699796952f;  // sqrt(512)
    xf[i] = v; xb[i] = (bf16)v;
}

__global__ void mask_k(const int* __restrict__ seq, float* __restrict__ maskf,
                       float* __restrict__ outmask)
{
    int i = blockIdx.x * blockDim.x + threadIdx.x;
    if (i >= BB * SS) return;
    bool pad = (seq[i] == 0);
    maskf[i]   = pad ? -1.0e9f : 0.0f;
    outmask[i] = pad ? 1.0f : 0.0f;
}

__global__ void cvt_bf16_k(const float* __restrict__ src, bf16* __restrict__ dst, size_t n)
{
    size_t i = (size_t)blockIdx.x * blockDim.x + threadIdx.x;
    size_t stride = (size_t)gridDim.x * blockDim.x;
    for (; i < n; i += stride) dst[i] = (bf16)src[i];
}

// ---------------------------------------------------------------------------
extern "C" void kernel_launch(void* const* d_in, const int* in_sizes, int n_in,
                              void* d_out, int out_size, void* d_ws, size_t ws_size,
                              hipStream_t stream)
{
    (void)in_sizes; (void)n_in; (void)out_size; (void)ws_size;

    const int*   seq       = (const int*)d_in[0];
    const int*   squares   = (const int*)d_in[1];
    const float* tok_emb   = (const float*)d_in[2];
    const float* dist_emb  = (const float*)d_in[3];
    const float* in_proj_w = (const float*)d_in[4];
    const float* in_proj_b = (const float*)d_in[5];
    const float* out_w     = (const float*)d_in[6];
    const float* out_b     = (const float*)d_in[7];
    const float* ff1_w     = (const float*)d_in[8];
    const float* ff1_b     = (const float*)d_in[9];
    const float* ff2_w     = (const float*)d_in[10];
    const float* ff2_b     = (const float*)d_in[11];
    const float* ln1_w     = (const float*)d_in[12];
    const float* ln1_b     = (const float*)d_in[13];
    const float* ln2_w     = (const float*)d_in[14];
    const float* ln2_b     = (const float*)d_in[15];
    const float* lnf_w     = (const float*)d_in[16];
    const float* lnf_b     = (const float*)d_in[17];

    char* ws = (char*)d_ws;
    auto alloc = [&](size_t bytes) -> char* {
        char* p = ws;
        ws += (bytes + 255) & ~(size_t)255;
        return p;
    };

    const size_t M = (size_t)BB * SS;  // 8192
    bf16*  wq    = (bf16*)alloc(sizeof(bf16) * LL * 3 * EE * EE);
    bf16*  wo    = (bf16*)alloc(sizeof(bf16) * LL * EE * EE);
    bf16*  w1    = (bf16*)alloc(sizeof(bf16) * LL * FF * EE);
    bf16*  w2    = (bf16*)alloc(sizeof(bf16) * LL * EE * FF);
    float* xf    = (float*)alloc(sizeof(float) * M * EE);
    bf16*  xb    = (bf16*)alloc(sizeof(bf16) * M * EE);
    bf16*  qkvb  = (bf16*)alloc(sizeof(bf16) * M * 3 * EE);
    bf16*  ctxb  = (bf16*)alloc(sizeof(bf16) * M * EE);
    float* tmp   = (float*)alloc(sizeof(float) * M * EE);
    bf16*  hb    = (bf16*)alloc(sizeof(bf16) * M * FF);
    float* maskf = (float*)alloc(sizeof(float) * BB * SS);

    cvt_bf16_k<<<2048, 256, 0, stream>>>(in_proj_w, wq, (size_t)LL * 3 * EE * EE);
    cvt_bf16_k<<<2048, 256, 0, stream>>>(out_w,     wo, (size_t)LL * EE * EE);
    cvt_bf16_k<<<2048, 256, 0, stream>>>(ff1_w,     w1, (size_t)LL * FF * EE);
    cvt_bf16_k<<<2048, 256, 0, stream>>>(ff2_w,     w2, (size_t)LL * EE * FF);

    embed_k<<<(unsigned)((M * EE + 255) / 256), 256, 0, stream>>>(seq, tok_emb, xf, xb);

    float* outmem  = (float*)d_out;
    float* outmask = outmem + (size_t)SS * BB * EE;
    mask_k<<<(BB * SS + 255) / 256, 256, 0, stream>>>(seq, maskf, outmask);

    for (int l = 0; l < LL; ++l) {
        // QKV projection -> bf16
        gemm_bf16_k<<<dim3(3 * EE / 128, (unsigned)(M / 64)), 128, 0, stream>>>(
            xb, wq + (size_t)l * 3 * EE * EE, in_proj_b + (size_t)l * 3 * EE,
            nullptr, qkvb, (int)M, 3 * EE, EE, 0);
        // Attention -> ctx bf16
        attn_flash_k<<<dim3(SS / 64, BB * HH), 128, 0, stream>>>(
            qkvb, squares, dist_emb, maskf, ctxb);
        // Out projection -> f32
        gemm_bf16_k<<<dim3(EE / 128, (unsigned)(M / 64)), 128, 0, stream>>>(
            ctxb, wo + (size_t)l * EE * EE, out_b + (size_t)l * EE,
            tmp, nullptr, (int)M, EE, EE, 0);
        // x = LN(x + attn_out)
        add_ln_k<<<(unsigned)M, 128, 0, stream>>>(
            xf, tmp, ln1_w + (size_t)l * EE, ln1_b + (size_t)l * EE, xf, xb);
        // FF1 + GELU -> bf16
        gemm_bf16_k<<<dim3(FF / 128, (unsigned)(M / 64)), 128, 0, stream>>>(
            xb, w1 + (size_t)l * FF * EE, ff1_b + (size_t)l * FF,
            nullptr, hb, (int)M, FF, EE, 1);
        // FF2 -> f32
        gemm_bf16_k<<<dim3(EE / 128, (unsigned)(M / 64)), 128, 0, stream>>>(
            hb, w2 + (size_t)l * EE * FF, ff2_b + (size_t)l * EE,
            tmp, nullptr, (int)M, EE, FF, 0);
        // x = LN(x + ff)
        add_ln_k<<<(unsigned)M, 128, 0, stream>>>(
            xf, tmp, ln2_w + (size_t)l * EE, ln2_b + (size_t)l * EE, xf, xb);
    }

    final_ln_k<<<(unsigned)M, 128, 0, stream>>>(xf, lnf_w, lnf_b, outmem);
}
